// DecoderWithAttention_4552665333779
// MI455X (gfx1250) — matrix operand order repaired and checked by reading, not compiled, not run on hardware
//
#include <hip/hip_runtime.h>
#include <hip/hip_bf16.h>
#include <math.h>

// ---------------------------------------------------------------------------
// Types for WMMA
// ---------------------------------------------------------------------------
typedef __bf16 bf16_t;
typedef __attribute__((ext_vector_type(16))) __bf16       v16bf;
typedef __attribute__((ext_vector_type(8)))  float        v8f;
typedef __attribute__((ext_vector_type(4)))  unsigned int u32x4;

// Problem constants (match the reference)
static constexpr int Bn  = 64;     // batch
static constexpr int Ln  = 49;     // encoder locations
static constexpr int Tn  = 20;     // timesteps
static constexpr int ENC = 2048;
static constexpr int Dn  = 512;    // decoder dim
static constexpr int An  = 512;    // attention dim
static constexpr int En  = 512;    // embed dim
static constexpr int Vn  = 10000;  // vocab

// ---------------------------------------------------------------------------
// f32 -> bf16 convert
// ---------------------------------------------------------------------------
__global__ void cvt_bf16_kernel(const float* __restrict__ in,
                                bf16_t* __restrict__ out, int n) {
  int i = blockIdx.x * blockDim.x + threadIdx.x;
  if (i < n) out[i] = (bf16_t)in[i];
}

// f32 (K x N row-major) -> bf16 (N x K row-major)  [transpose + convert]
__global__ void transpose_cvt_kernel(const float* __restrict__ in,
                                     bf16_t* __restrict__ out, int K, int N) {
  int i = blockIdx.x * blockDim.x + threadIdx.x;
  if (i >= K * N) return;
  int n = i / K, k = i % K;           // output is N x K, contiguous in k
  out[i] = (bf16_t)in[k * N + n];
}

// embedding gather: emb_bf[b,t,e] = bf16(emb_table[captions[b,t], e])
__global__ void gather_emb_kernel(const float* __restrict__ emb_table,
                                  const int* __restrict__ captions,
                                  bf16_t* __restrict__ out) {
  int i = blockIdx.x * blockDim.x + threadIdx.x;
  int total = Bn * Tn * En;
  if (i >= total) return;
  int e  = i % En;
  int bt = i / En;            // b*Tn + t
  int tok = captions[bt];
  out[i] = (bf16_t)emb_table[(long)tok * En + e];
}

// b_gates = b_ih + b_hh
__global__ void add_vec_kernel(const float* a, const float* b, float* out, int n) {
  int i = blockIdx.x * blockDim.x + threadIdx.x;
  if (i < n) out[i] = a[i] + b[i];
}

// mean over L: mean[b,e] = (1/L) * sum_l enc[b,l,e]
__global__ void mean_enc_kernel(const float* __restrict__ enc,
                                float* __restrict__ mean) {
  int i = blockIdx.x * blockDim.x + threadIdx.x;   // Bn*ENC
  if (i >= Bn * ENC) return;
  int b = i / ENC, e = i % ENC;
  const float* p = enc + (long)b * Ln * ENC + e;
  float s = 0.f;
  #pragma unroll 7
  for (int l = 0; l < Ln; ++l) s += p[(long)l * ENC];
  mean[i] = s * (1.0f / (float)Ln);
}

// h/c init: out[b,n] = bias[n] + sum_k mean[b,k] * W[k*D + n]   (W is ENC x D)
__global__ void init_hc_kernel(const float* __restrict__ mean,
                               const float* __restrict__ W,
                               const float* __restrict__ bias,
                               float* __restrict__ out,
                               bf16_t* __restrict__ out_bf) {
  int i = blockIdx.x * blockDim.x + threadIdx.x;   // Bn*Dn
  if (i >= Bn * Dn) return;
  int b = i >> 9, n = i & (Dn - 1);
  const float* mb = mean + (long)b * ENC;
  float acc = bias[n];
  for (int k = 0; k < ENC; ++k) acc = fmaf(mb[k], W[(long)k * Dn + n], acc);
  out[i] = acc;
  if (out_bf) out_bf[i] = (bf16_t)acc;
}

// ---------------------------------------------------------------------------
// WMMA bf16 GEMM:  C[M x N] (+)= A[M x K] * B^T  (B stored N x K row-major)
//   - one 16x16 output tile per wave (wave32), K consumed 32/instruction
//   - fragment loads are contiguous 16B per lane per the CDNA5 ISA layouts
//   - accumulate==0: init from bias[n]; accumulate==1: read-modify-write C
// ---------------------------------------------------------------------------
__global__ __launch_bounds__(256)
void wmma_gemm_bt_kernel(const bf16_t* __restrict__ A, int lda,
                         const bf16_t* __restrict__ B, int ldb,
                         float* __restrict__ C, long ldc,
                         const float* __restrict__ bias,
                         int M, int N, int K, int accumulate) {
  const int lane  = threadIdx.x & 31;
  const int wave  = (blockIdx.x * blockDim.x + threadIdx.x) >> 5;
  const int tilesN = N >> 4;
  const int tilesM = (M + 15) >> 4;
  if (wave >= tilesM * tilesN) return;          // wave-uniform: EXEC stays all-1s
  const int tm = wave / tilesN;
  const int tn = wave - tm * tilesN;
  const int m0 = tm << 4, n0 = tn << 4;

  const int half = lane >> 4;      // 0: lanes 0-15, 1: lanes 16-31
  const int l16  = lane & 15;

  // --- accumulator init (C/D layout: VGPR r -> M = r + 8*half, N = l16) ---
  const int  cn    = n0 + l16;
  const int  mbase = m0 + (half << 3);
  v8f acc;
  if (accumulate) {
    #pragma unroll
    for (int r = 0; r < 8; ++r) acc[r] = C[(long)(mbase + r) * ldc + cn];
  } else {
    const float bv = bias ? bias[cn] : 0.0f;
    #pragma unroll
    for (int r = 0; r < 8; ++r) acc[r] = bv;
  }

  // A fragment (16x32 bf16): lane row = m0+l16; K bytes split {0..7,16..23}/{8..15,24..31}
  const bf16_t* aRow = A + (long)(m0 + l16) * lda + (half << 3);
  // B fragment (32x16 bf16), B stored N x K: lane col = n0+l16; K split {0..7,16..23}/{8..15,24..31}
  const bf16_t* bRow = B + (long)(n0 + l16) * ldb + (half << 3);

  union Frag { u32x4 u[2]; v16bf v; };

  #pragma unroll 2
  for (int k0 = 0; k0 < K; k0 += 32) {
    Frag fa, fb;
    fa.u[0] = *(const u32x4*)(aRow + k0);
    fa.u[1] = *(const u32x4*)(aRow + k0 + 16);
    fb.u[0] = *(const u32x4*)(bRow + k0);
    fb.u[1] = *(const u32x4*)(bRow + k0 + 16);
    // speculative prefetch of next K tiles (weights are L2 resident)
    __builtin_prefetch(aRow + k0 + 64, 0, 1);
    __builtin_prefetch(bRow + k0 + 64, 0, 1);
    acc = __builtin_amdgcn_wmma_f32_16x16x32_bf16(
        /*neg_a=*/false, fa.v, /*neg_b=*/false, fb.v,
        /*c_mod=*/(short)0, acc, /*reuse_a=*/false, /*reuse_b=*/false);
  }

  #pragma unroll
  for (int r = 0; r < 8; ++r) C[(long)(mbase + r) * ldc + cn] = acc[r];
}

// ---------------------------------------------------------------------------
// Fused additive attention for one timestep. One block (256 thr) per batch b.
//   dec_proj = h @ W_dec_att + b_dec_att
//   scores[l] = w_full . relu(enc_proj[b,l,:] + dec_proj) + b_full
//   alpha = softmax_l(scores);  context_bf[b,:] = sum_l alpha_l * enc[b,l,:]
// ---------------------------------------------------------------------------
__global__ __launch_bounds__(256)
void attention_kernel(const float* __restrict__ h,          // (B,D)
                      const float* __restrict__ W_dec_att,  // (D,A)
                      const float* __restrict__ b_dec_att,  // (A)
                      const float* __restrict__ w_full,     // (A)
                      const float* __restrict__ b_full,     // scalar
                      const float* __restrict__ enc_proj,   // (B,L,A)
                      const float* __restrict__ enc,        // (B,L,ENC)
                      float* __restrict__ alphas_out,       // -> d_out alphas + t*L
                      bf16_t* __restrict__ context_bf) {    // (B,ENC)
  __shared__ float dec_s[An];
  __shared__ float score_s[64];
  __shared__ float alpha_s[64];

  const int b   = blockIdx.x;
  const int tid = threadIdx.x;
  const int wid = tid >> 5, lane = tid & 31;

  // Phase 1: dec_proj (coalesced over n across lanes)
  for (int n = tid; n < An; n += 256) {
    float acc = b_dec_att[n];
    const float* hb = h + (long)b * Dn;
    for (int k = 0; k < Dn; ++k) acc = fmaf(hb[k], W_dec_att[(long)k * An + n], acc);
    dec_s[n] = acc;
  }
  __syncthreads();

  // Phase 2: scores; one location per wave, lanes stride the A dim
  for (int l = wid; l < Ln; l += 8) {
    const float* ep = enc_proj + ((long)b * Ln + l) * An;
    float part = 0.f;
    for (int a = lane; a < An; a += 32) {
      float v = ep[a] + dec_s[a];
      v = fmaxf(v, 0.0f);
      part = fmaf(v, w_full[a], part);
    }
    #pragma unroll
    for (int off = 16; off > 0; off >>= 1) part += __shfl_xor(part, off, 32);
    if (lane == 0) score_s[l] = part + b_full[0];
  }
  __syncthreads();

  // Phase 3: softmax over L=49 (tiny; single thread)
  if (tid == 0) {
    float mx = score_s[0];
    for (int l = 1; l < Ln; ++l) mx = fmaxf(mx, score_s[l]);
    float sum = 0.f;
    for (int l = 0; l < Ln; ++l) { float e = __expf(score_s[l] - mx); alpha_s[l] = e; sum += e; }
    float inv = 1.0f / sum;
    for (int l = 0; l < Ln; ++l) alpha_s[l] *= inv;
  }
  __syncthreads();

  // write alphas to d_out
  for (int l = tid; l < Ln; l += 256)
    alphas_out[(long)b * Tn * Ln + l] = alpha_s[l];

  // Phase 4: context (coalesced over e), emit bf16 for the gates GEMM
  for (int e = tid; e < ENC; e += 256) {
    const float* eb = enc + (long)b * Ln * ENC + e;
    float ctx = 0.f;
    #pragma unroll 7
    for (int l = 0; l < Ln; ++l) ctx = fmaf(alpha_s[l], eb[(long)l * ENC], ctx);
    context_bf[(long)b * ENC + e] = (bf16_t)ctx;
  }
}

// ---------------------------------------------------------------------------
// LSTM pointwise: gates (B,4D) order i,f,g,o -> h_new, c_new
// ---------------------------------------------------------------------------
__global__ void lstm_pointwise_kernel(const float* __restrict__ gates,
                                      float* __restrict__ c,
                                      float* __restrict__ h,
                                      bf16_t* __restrict__ h_bf) {
  int idx = blockIdx.x * blockDim.x + threadIdx.x;     // Bn*Dn
  if (idx >= Bn * Dn) return;
  int b = idx >> 9, j = idx & (Dn - 1);
  const float* g = gates + (long)b * (4 * Dn);
  float gi = g[j];
  float gf = g[j + Dn];
  float gg = g[j + 2 * Dn];
  float go = g[j + 3 * Dn];
  float i_ = 1.0f / (1.0f + __expf(-gi));
  float f_ = 1.0f / (1.0f + __expf(-gf));
  float o_ = 1.0f / (1.0f + __expf(-go));
  float g_ = tanhf(gg);
  float cn = f_ * c[idx] + i_ * g_;
  float hn = o_ * tanhf(cn);
  c[idx] = cn;
  h[idx] = hn;
  h_bf[idx] = (bf16_t)hn;
}

// ---------------------------------------------------------------------------
// Host-side orchestration
// ---------------------------------------------------------------------------
static inline size_t align256(size_t x) { return (x + 255) & ~(size_t)255; }

extern "C" void kernel_launch(void* const* d_in, const int* in_sizes, int n_in,
                              void* d_out, int out_size, void* d_ws, size_t ws_size,
                              hipStream_t stream) {
  (void)in_sizes; (void)n_in; (void)out_size; (void)ws_size;

  const float* encoder_out = (const float*)d_in[0];   // (B,L,ENC)
  const int*   captions    = (const int*)  d_in[1];   // (B,T)
  const float* emb_table   = (const float*)d_in[2];   // (V,E)
  const float* W_enc_att   = (const float*)d_in[3];   // (ENC,A)
  const float* b_enc_att   = (const float*)d_in[4];
  const float* W_dec_att   = (const float*)d_in[5];   // (D,A)
  const float* b_dec_att   = (const float*)d_in[6];
  const float* w_full      = (const float*)d_in[7];
  const float* b_full      = (const float*)d_in[8];   // scalar
  const float* W_ih        = (const float*)d_in[9];   // (4D, E+ENC)
  const float* b_ih        = (const float*)d_in[10];
  const float* W_hh        = (const float*)d_in[11];  // (4D, D)
  const float* b_hh        = (const float*)d_in[12];
  const float* W_init_h    = (const float*)d_in[13];  // (ENC, D)
  const float* b_init_h    = (const float*)d_in[14];
  const float* W_init_c    = (const float*)d_in[15];
  const float* b_init_c    = (const float*)d_in[16];
  const float* W_out       = (const float*)d_in[17];  // (D, V)
  const float* b_out       = (const float*)d_in[18];

  float* logits = (float*)d_out;                                  // (B,T,V)
  float* alphas = (float*)d_out + (size_t)Bn * Tn * Vn;           // (B,T,L)

  // ---- workspace bump allocator ----
  char* ws = (char*)d_ws;
  size_t off = 0;
  auto alloc = [&](size_t bytes) -> void* {
    void* p = ws + off; off = align256(off + bytes); return p;
  };
  bf16_t* enc_bf     = (bf16_t*)alloc((size_t)Bn * Ln * ENC * 2);       // (B*L, ENC)
  bf16_t* WencT_bf   = (bf16_t*)alloc((size_t)An * ENC * 2);            // (A, ENC)
  bf16_t* Wih_bf     = (bf16_t*)alloc((size_t)(4 * Dn) * (En + ENC) * 2);
  bf16_t* Whh_bf     = (bf16_t*)alloc((size_t)(4 * Dn) * Dn * 2);
  bf16_t* WoutT_bf   = (bf16_t*)alloc((size_t)Vn * Dn * 2);             // (V, D)
  bf16_t* emb_bf     = (bf16_t*)alloc((size_t)Bn * Tn * En * 2);
  float*  b_gates    = (float*) alloc((size_t)(4 * Dn) * 4);
  float*  mean_enc   = (float*) alloc((size_t)Bn * ENC * 4);
  float*  h_buf      = (float*) alloc((size_t)Bn * Dn * 4);
  float*  c_buf      = (float*) alloc((size_t)Bn * Dn * 4);
  bf16_t* h_bf       = (bf16_t*)alloc((size_t)Bn * Dn * 2);
  float*  enc_proj   = (float*) alloc((size_t)Bn * Ln * An * 4);
  bf16_t* ctx_bf     = (bf16_t*)alloc((size_t)Bn * ENC * 2);
  float*  gates      = (float*) alloc((size_t)Bn * 4 * Dn * 4);

  auto grid1 = [](int n) { return (n + 255) / 256; };
  auto gemm  = [&](const bf16_t* A, int lda, const bf16_t* Bm, int ldb,
                   float* C, long ldc, const float* bias,
                   int M, int N, int K, int accumulate) {
    int tiles = ((M + 15) / 16) * (N / 16);
    int grid  = (tiles * 32 + 255) / 256;
    wmma_gemm_bt_kernel<<<grid, 256, 0, stream>>>(A, lda, Bm, ldb, C, ldc, bias,
                                                  M, N, K, accumulate);
  };

  // ---- one-time prep (re-run every call for determinism; all L2-resident) ----
  cvt_bf16_kernel<<<grid1(Bn * Ln * ENC), 256, 0, stream>>>(encoder_out, enc_bf, Bn * Ln * ENC);
  cvt_bf16_kernel<<<grid1(4 * Dn * (En + ENC)), 256, 0, stream>>>(W_ih, Wih_bf, 4 * Dn * (En + ENC));
  cvt_bf16_kernel<<<grid1(4 * Dn * Dn), 256, 0, stream>>>(W_hh, Whh_bf, 4 * Dn * Dn);
  transpose_cvt_kernel<<<grid1(ENC * An), 256, 0, stream>>>(W_enc_att, WencT_bf, ENC, An);
  transpose_cvt_kernel<<<grid1(Dn * Vn), 256, 0, stream>>>(W_out, WoutT_bf, Dn, Vn);
  gather_emb_kernel<<<grid1(Bn * Tn * En), 256, 0, stream>>>(emb_table, captions, emb_bf);
  add_vec_kernel<<<grid1(4 * Dn), 256, 0, stream>>>(b_ih, b_hh, b_gates, 4 * Dn);
  mean_enc_kernel<<<grid1(Bn * ENC), 256, 0, stream>>>(encoder_out, mean_enc);
  init_hc_kernel<<<grid1(Bn * Dn), 256, 0, stream>>>(mean_enc, W_init_h, b_init_h, h_buf, h_bf);
  init_hc_kernel<<<grid1(Bn * Dn), 256, 0, stream>>>(mean_enc, W_init_c, b_init_c, c_buf, nullptr);

  // enc_proj = encoder_out @ W_enc_att + b_enc_att : (B*L, A)
  gemm(enc_bf, ENC, WencT_bf, ENC, enc_proj, An, b_enc_att,
       Bn * Ln, An, ENC, /*accumulate=*/0);

  // ---- sequential decode over T steps ----
  for (int t = 0; t < Tn; ++t) {
    attention_kernel<<<Bn, 256, 0, stream>>>(h_buf, W_dec_att, b_dec_att, w_full,
                                             b_full, enc_proj, encoder_out,
                                             alphas + (size_t)t * Ln, ctx_bf);

    // gates = emb_t @ Wih[:, :E]^T + context @ Wih[:, E:]^T + h @ Whh^T + (b_ih+b_hh)
    gemm(emb_bf + (size_t)t * En, Tn * En, Wih_bf, En + ENC,
         gates, 4 * Dn, b_gates, Bn, 4 * Dn, En, /*accumulate=*/0);
    gemm(ctx_bf, ENC, Wih_bf + En, En + ENC,
         gates, 4 * Dn, nullptr, Bn, 4 * Dn, ENC, /*accumulate=*/1);
    gemm(h_bf, Dn, Whh_bf, Dn,
         gates, 4 * Dn, nullptr, Bn, 4 * Dn, Dn, /*accumulate=*/1);

    lstm_pointwise_kernel<<<grid1(Bn * Dn), 256, 0, stream>>>(gates, c_buf, h_buf, h_bf);

    // logits[:, t, :] = h_new @ W_out + b_out ; row stride T*V, col offset t*V
    gemm(h_bf, Dn, WoutT_bf, Dn,
         logits + (size_t)t * Vn, (long)Tn * Vn, b_out, Bn, Vn, Dn, /*accumulate=*/0);
  }
}
